// Block_76510547411282
// MI455X (gfx1250) — matrix operand order repaired and checked by reading, not compiled, not run on hardware
//
#include <hip/hip_runtime.h>
#include <hip/hip_bf16.h>
#include <math.h>

// ---------------------------------------------------------------------------
// Problem constants (match reference)
// ---------------------------------------------------------------------------
#define BB   2
#define NN   2048
#define CC   1024
#define HH   16
#define DD   64
#define DFF  4096
#define MM   (BB * NN)          // 4096 token rows

typedef __bf16 bf16_t;
typedef __bf16 bf16x8  __attribute__((ext_vector_type(8)));
typedef __bf16 bf16x16 __attribute__((ext_vector_type(16)));
typedef float  f32x4   __attribute__((ext_vector_type(4)));
typedef float  v8f     __attribute__((ext_vector_type(8)));

// ---------------------------------------------------------------------------
// WMMA fragment loaders (16x16x32 bf16, wave32; contiguous k order chosen here, same for A and B)
//
// A (16x32, MxK): lanes 0-15 -> rows 0-15, K=0..15;
//                 lanes 16-31 -> rows 0-15, K=16..31.
// B (32x16, KxN): lanes 0-15 -> N=0..15 with K=0..15 (16 halves ascending);
//                 lanes 16-31 -> N=0..15 with K=16..31.
// B is read from memory stored as Bmem[n][k] (row-major NxK, i.e. A @ B^T),
// so every lane's halves are contiguous 16B/32B chunks in memory.
// ---------------------------------------------------------------------------
__device__ __forceinline__ bf16x16 load_frag_A(const bf16_t* __restrict__ A,
                                               int lda, int m0, int k0, int lane) {
  int row = m0 + (lane & 15);
  int kb  = k0 + ((lane >> 4) << 4);
  const bf16_t* p = A + (long long)row * lda + kb;
  union { bf16x16 v; bf16x8 h[2]; } u;
  u.h[0] = *(const bf16x8*)(p);
  u.h[1] = *(const bf16x8*)(p + 8);
  return u.v;
}

__device__ __forceinline__ bf16x16 load_frag_A(const float* __restrict__ A,
                                               int lda, int m0, int k0, int lane) {
  int row = m0 + (lane & 15);
  int kb  = k0 + ((lane >> 4) << 4);
  const float* p = A + (long long)row * lda + kb;
  float t[16];
  *(f32x4*)&t[0]  = *(const f32x4*)(p);
  *(f32x4*)&t[4]  = *(const f32x4*)(p + 4);
  *(f32x4*)&t[8]  = *(const f32x4*)(p + 8);
  *(f32x4*)&t[12] = *(const f32x4*)(p + 12);
  bf16x16 v;
#pragma unroll
  for (int i = 0; i < 16; ++i) v[i] = (bf16_t)t[i];
  return v;
}

__device__ __forceinline__ bf16x16 load_frag_B(const bf16_t* __restrict__ Bm,
                                               int ldb, int n0, int k0, int lane) {
  int n  = n0 + (lane & 15);
  int kb = k0 + ((lane >> 4) << 4);
  const bf16_t* p = Bm + (long long)n * ldb + kb;
  union { bf16x16 v; bf16x8 h[2]; } u;
  u.h[0] = *(const bf16x8*)(p);
  u.h[1] = *(const bf16x8*)(p + 8);
  return u.v;
}

__device__ __forceinline__ bf16x16 load_frag_B(const float* __restrict__ Bm,
                                               int ldb, int n0, int k0, int lane) {
  int n  = n0 + (lane & 15);
  int kb = k0 + ((lane >> 4) << 4);
  const float* p = Bm + (long long)n * ldb + kb;
  float t[16];
  *(f32x4*)&t[0]  = *(const f32x4*)(p);
  *(f32x4*)&t[4]  = *(const f32x4*)(p + 4);
  *(f32x4*)&t[8]  = *(const f32x4*)(p + 8);
  *(f32x4*)&t[12] = *(const f32x4*)(p + 12);
  bf16x16 v;
#pragma unroll
  for (int i = 0; i < 16; ++i) v[i] = (bf16_t)t[i];
  return v;
}

__device__ __forceinline__ v8f wmma_bf16(bf16x16 a, bf16x16 b, v8f c) {
  return __builtin_amdgcn_wmma_f32_16x16x32_bf16(false, a, false, b,
                                                 (short)0, c, false, false);
}

// ---------------------------------------------------------------------------
// Generic batched WMMA GEMM:   C = epilogue( A[MxK] @ Bmem[NxK]^T )
// Each wave computes one 32x64 tile: 2x4 WMMA accumulators, A-frag reuse x4,
// B-frag reuse x2 (8 v_wmma per 6 fragment loads per K-step).
// blockIdx.z = b*Hh + h with per-operand (b,h) offset decomposition.
// EPI: 0 = plain (+bias/residual per template flags)
//      1 = attn scores: v*scale, mask[col]==0 -> -inf
//      2 = bias + exact GELU
// ---------------------------------------------------------------------------
template <typename TA, typename TB, typename TC, int EPI, bool HAS_BIAS, bool HAS_RES>
__global__ __launch_bounds__(256) void gemm_wmma(
    int M, int N, int K, int Hh,
    const TA* __restrict__ A, int lda, long long aOffB, long long aOffH,
    const TB* __restrict__ Bm, int ldb, long long bOffB, long long bOffH,
    TC* __restrict__ Cc, int ldc, long long cOffB, long long cOffH,
    const float* __restrict__ bias,
    const float* __restrict__ residual, int ldr,
    float scale, const int* __restrict__ maskp, int maskLen) {
  const int z  = blockIdx.z;
  const int zb = z / Hh;
  const int zh = z % Hh;
  A  += zb * aOffB + zh * aOffH;
  Bm += zb * bOffB + zh * bOffH;
  Cc += zb * cOffB + zh * cOffH;

  const int tilesN = N >> 6;                     // 64-wide wave tiles
  const int tilesM = M >> 5;                     // 32-tall wave tiles
  const int wave   = blockIdx.x * 8 + (threadIdx.x >> 5);
  if (wave >= tilesM * tilesN) return;           // wave-uniform: EXEC stays full
  const int tm   = wave / tilesN;
  const int tn   = wave % tilesN;
  const int lane = threadIdx.x & 31;
  const int m0   = tm << 5;
  const int n0   = tn << 6;

  v8f acc[2][4] = {};

  for (int k0 = 0; k0 < K; k0 += 32) {
    // prefetch 2 K-steps ahead (speculative global_prefetch_b8; OOB is dropped)
    {
      int pk = k0 + 64;
      __builtin_prefetch(A + (long long)(m0 + (lane & 15)) * lda + pk, 0, 0);
      __builtin_prefetch(Bm + (long long)(n0 + lane) * ldb + pk, 0, 0);
      __builtin_prefetch(Bm + (long long)(n0 + 32 + lane) * ldb + pk, 0, 0);
    }
    bf16x16 a0 = load_frag_A(A, lda, m0,      k0, lane);
    bf16x16 a1 = load_frag_A(A, lda, m0 + 16, k0, lane);
    bf16x16 b0 = load_frag_B(Bm, ldb, n0,      k0, lane);
    bf16x16 b1 = load_frag_B(Bm, ldb, n0 + 16, k0, lane);
    bf16x16 b2 = load_frag_B(Bm, ldb, n0 + 32, k0, lane);
    bf16x16 b3 = load_frag_B(Bm, ldb, n0 + 48, k0, lane);
    acc[0][0] = wmma_bf16(a0, b0, acc[0][0]);
    acc[0][1] = wmma_bf16(a0, b1, acc[0][1]);
    acc[0][2] = wmma_bf16(a0, b2, acc[0][2]);
    acc[0][3] = wmma_bf16(a0, b3, acc[0][3]);
    acc[1][0] = wmma_bf16(a1, b0, acc[1][0]);
    acc[1][1] = wmma_bf16(a1, b1, acc[1][1]);
    acc[1][2] = wmma_bf16(a1, b2, acc[1][2]);
    acc[1][3] = wmma_bf16(a1, b3, acc[1][3]);
  }

  // Epilogue. C/D layout: VGPR r -> (M = hi*8 + r, N = lane&15).
  const int hi = lane >> 4;
  const int ln = lane & 15;
#pragma unroll
  for (int tj = 0; tj < 4; ++tj) {
    const int col = n0 + tj * 16 + ln;
    float colBias = 0.f;
    bool  keep    = true;
    if (EPI == 1) keep = (maskp[zb * maskLen + col] != 0);
    if (HAS_BIAS) colBias = bias[col];
#pragma unroll
    for (int ti = 0; ti < 2; ++ti) {
#pragma unroll
      for (int r = 0; r < 8; ++r) {
        int   row = m0 + ti * 16 + (hi << 3) + r;
        float v   = acc[ti][tj][r];
        if (EPI == 1) {
          v = keep ? v * scale : -__builtin_inff();
        } else {
          if (HAS_BIAS) v += colBias;
          if (EPI == 2) v = 0.5f * v * (1.0f + erff(v * 0.70710678118654752f));
          if (HAS_RES)  v += residual[(long long)row * ldr + col];
        }
        Cc[(long long)row * ldc + col] = (TC)v;
      }
    }
  }
}

// ---------------------------------------------------------------------------
// LayerNorm over last dim (C=1024), fp32 in -> bf16 out. One block per row.
// ---------------------------------------------------------------------------
__global__ __launch_bounds__(256) void layernorm_bf16(
    const float* __restrict__ src, const float* __restrict__ g,
    const float* __restrict__ bvec, bf16_t* __restrict__ dst, int Cdim) {
  const int  row = blockIdx.x;
  const int  tid = threadIdx.x;
  const float* p = src + (long long)row * Cdim;
  __shared__ float red[256];

  float s = 0.f;
  for (int i = tid; i < Cdim; i += 256) s += p[i];
  red[tid] = s; __syncthreads();
  for (int o = 128; o > 0; o >>= 1) {
    if (tid < o) red[tid] += red[tid + o];
    __syncthreads();
  }
  const float mu = red[0] / (float)Cdim;
  __syncthreads();

  float v = 0.f;
  for (int i = tid; i < Cdim; i += 256) { float d = p[i] - mu; v += d * d; }
  red[tid] = v; __syncthreads();
  for (int o = 128; o > 0; o >>= 1) {
    if (tid < o) red[tid] += red[tid + o];
    __syncthreads();
  }
  const float inv = rsqrtf(red[0] / (float)Cdim + 1e-5f);

  bf16_t* q = dst + (long long)row * Cdim;
  for (int i = tid; i < Cdim; i += 256)
    q[i] = (bf16_t)((p[i] - mu) * inv * g[i] + bvec[i]);
}

// ---------------------------------------------------------------------------
// Row softmax, in place on fp32 attn scores. One block per row of length N.
// ---------------------------------------------------------------------------
__global__ __launch_bounds__(256) void softmax_rows(float* __restrict__ attn,
                                                    int ncols) {
  const long long row = blockIdx.x;
  const int tid = threadIdx.x;
  float* p = attn + row * (long long)ncols;
  __shared__ float red[256];

  float mx = -__builtin_inff();
  for (int i = tid; i < ncols; i += 256) mx = fmaxf(mx, p[i]);
  red[tid] = mx; __syncthreads();
  for (int o = 128; o > 0; o >>= 1) {
    if (tid < o) red[tid] = fmaxf(red[tid], red[tid + o]);
    __syncthreads();
  }
  mx = red[0]; __syncthreads();

  float s = 0.f;
  for (int i = tid; i < ncols; i += 256) {
    float e = __expf(p[i] - mx);
    p[i] = e;
    s += e;
  }
  red[tid] = s; __syncthreads();
  for (int o = 128; o > 0; o >>= 1) {
    if (tid < o) red[tid] += red[tid + o];
    __syncthreads();
  }
  const float inv = 1.0f / red[0];
  for (int i = tid; i < ncols; i += 256) p[i] *= inv;
}

// ---------------------------------------------------------------------------
// Transpose V out of the qkv buffer: Vt[b,h,d,j] = qkv[b*N+j, 2C + h*D + d]
// ---------------------------------------------------------------------------
__global__ __launch_bounds__(256) void transpose_v(
    const bf16_t* __restrict__ qkv, bf16_t* __restrict__ vt) {
  const long long total = (long long)BB * HH * DD * NN;
  long long idx = (long long)blockIdx.x * 256 + threadIdx.x;
  if (idx >= total) return;
  int j = (int)(idx & (NN - 1));
  long long t = idx >> 11;            // / NN
  int d = (int)(t & (DD - 1));  t >>= 6;
  int h = (int)(t & (HH - 1));  t >>= 4;
  int b = (int)t;
  vt[idx] = qkv[((long long)(b * NN + j)) * (3 * CC) + 2 * CC + h * DD + d];
}

// ---------------------------------------------------------------------------
static inline int gemm_blocks(int M, int N) {
  int tiles = (M / 32) * (N / 64);   // 32x64 wave tiles
  return (tiles + 7) / 8;            // 8 waves per 256-thread block
}

extern "C" void kernel_launch(void* const* d_in, const int* in_sizes, int n_in,
                              void* d_out, int out_size, void* d_ws, size_t ws_size,
                              hipStream_t stream) {
  const float* x      = (const float*)d_in[0];
  const int*   mask   = (const int*)d_in[1];
  const float* qkv_w  = (const float*)d_in[2];
  const float* proj_w = (const float*)d_in[3];
  const float* proj_b = (const float*)d_in[4];
  const float* ln1_g  = (const float*)d_in[5];
  const float* ln1_b  = (const float*)d_in[6];
  const float* ln2_g  = (const float*)d_in[7];
  const float* ln2_b  = (const float*)d_in[8];
  const float* fc1_w  = (const float*)d_in[9];
  const float* fc1_b  = (const float*)d_in[10];
  const float* fc2_w  = (const float*)d_in[11];
  const float* fc2_b  = (const float*)d_in[12];

  float* out_x    = (float*)d_out;
  float* out_attn = out_x + (size_t)MM * CC;   // [B,H,N,N] after [B,N,C]

  // Workspace layout (bytes), peak ~72 MB with reuse:
  char*   ws   = (char*)d_ws;
  bf16_t* xn   = (bf16_t*)(ws);                         //  8 MB [0,8M)   ln1 out / ln2 out
  bf16_t* qkvb = (bf16_t*)(ws + (8ull  << 20));         // 24 MB [8,32M)  qkv bf16
  bf16_t* vt   = (bf16_t*)(ws + (32ull << 20));         // 16 MB [32,48M) V transposed
  bf16_t* obuf = (bf16_t*)(ws + (48ull << 20));         //  8 MB [48,56M) attn out (pre-proj)
  float*  x1   = (float*) (ws + (56ull << 20));         // 16 MB [56,72M) residual-1 result
  bf16_t* h1   = (bf16_t*)(ws + (8ull  << 20));         // 32 MB reuse [8,40M) (qkv/vt dead)
  bf16_t* hln  = xn;                                    // reuse (xn dead after qkv gemm)

  const float scale = 0.125f;  // D^-0.5, D=64

  // 1) xn = LN1(x) -> bf16
  layernorm_bf16<<<MM, 256, 0, stream>>>(x, ln1_g, ln1_b, xn, CC);

  // 2) qkv = xn @ qkv_w^T -> bf16  [4096 x 3072]
  gemm_wmma<bf16_t, float, bf16_t, 0, false, false>
      <<<dim3(gemm_blocks(MM, 3 * CC), 1, 1), 256, 0, stream>>>(
      MM, 3 * CC, CC, 1,
      xn, CC, 0, 0,
      qkv_w, CC, 0, 0,
      qkvb, 3 * CC, 0, 0,
      nullptr, nullptr, 0, 1.0f, nullptr, 0);

  // 3) S = scale * Q @ K^T with mask -> fp32 scores in out_attn (batched B*H)
  gemm_wmma<bf16_t, bf16_t, float, 1, false, false>
      <<<dim3(gemm_blocks(NN, NN), 1, BB * HH), 256, 0, stream>>>(
      NN, NN, DD, HH,
      qkvb,      3 * CC, (long long)NN * 3 * CC, DD,                 // Q
      qkvb + CC, 3 * CC, (long long)NN * 3 * CC, DD,                 // K
      out_attn,  NN,     (long long)HH * NN * NN, (long long)NN * NN,
      nullptr, nullptr, 0, scale, mask, NN);

  // 4) softmax rows in place (this is the returned attn)
  softmax_rows<<<BB * HH * NN, 256, 0, stream>>>(out_attn, NN);

  // 5) Vt[b,h,d,j] from qkv buffer
  {
    long long total = (long long)BB * HH * DD * NN;
    transpose_v<<<(unsigned)((total + 255) / 256), 256, 0, stream>>>(qkvb, vt);
  }

  // 6) O = P @ V (A = fp32 softmaxed P, inline bf16 cvt) -> obuf [b,n,h,d] bf16
  gemm_wmma<float, bf16_t, bf16_t, 0, false, false>
      <<<dim3(gemm_blocks(NN, DD), 1, BB * HH), 256, 0, stream>>>(
      NN, DD, NN, HH,
      out_attn, NN,  (long long)HH * NN * NN, (long long)NN * NN,
      vt,       NN,  (long long)HH * DD * NN, (long long)DD * NN,
      obuf,     CC,  (long long)NN * CC,      (long long)DD,
      nullptr, nullptr, 0, 1.0f, nullptr, 0);

  // 7) x1 = x + obuf @ proj_w^T + proj_b   (fp32)
  gemm_wmma<bf16_t, float, float, 0, true, true>
      <<<dim3(gemm_blocks(MM, CC), 1, 1), 256, 0, stream>>>(
      MM, CC, CC, 1,
      obuf, CC, 0, 0,
      proj_w, CC, 0, 0,
      x1, CC, 0, 0,
      proj_b, x, CC, 1.0f, nullptr, 0);

  // 8) hln = LN2(x1) -> bf16
  layernorm_bf16<<<MM, 256, 0, stream>>>(x1, ln2_g, ln2_b, hln, CC);

  // 9) h1 = GELU(hln @ fc1_w^T + fc1_b) -> bf16 [4096 x 4096]
  gemm_wmma<bf16_t, float, bf16_t, 2, true, false>
      <<<dim3(gemm_blocks(MM, DFF), 1, 1), 256, 0, stream>>>(
      MM, DFF, CC, 1,
      hln, CC, 0, 0,
      fc1_w, CC, 0, 0,
      h1, DFF, 0, 0,
      fc1_b, nullptr, 0, 1.0f, nullptr, 0);

  // 10) out_x = x1 + h1 @ fc2_w^T + fc2_b  (fp32)
  gemm_wmma<bf16_t, float, float, 0, true, true>
      <<<dim3(gemm_blocks(MM, CC), 1, 1), 256, 0, stream>>>(
      MM, CC, DFF, 1,
      h1, DFF, 0, 0,
      fc2_w, DFF, 0, 0,
      out_x, CC, 0, 0,
      fc2_b, x1, CC, 1.0f, nullptr, 0);
}
